// SpaceTimeLookTable_80264348827718
// MI455X (gfx1250) — matrix operand order repaired and checked by reading, not compiled, not run on hardware
//
#include <hip/hip_runtime.h>

typedef unsigned short u16;
typedef __attribute__((ext_vector_type(4)))  unsigned int u32x4;
typedef __attribute__((ext_vector_type(8)))  float        v8f;
typedef __attribute__((ext_vector_type(16))) __bf16       v16bf;

union BF16Tile { u32x4 q[2]; v16bf v; };

#define NPTS   32768
#define KTOT   2336      // TOTAL features
#define NOUT1  4672      // 2*TOTAL
#define KT1    73        // KTOT/32  (K tiles of main GEMM)
#define NT1    292       // NOUT1/16 (N tiles of vw1)
#define JT     146       // KTOT/16  (v column tiles)
#define MT     64        // rows per block
#define EPS    1e-8f

// ---------------------------------------------------------------- helpers
__device__ __forceinline__ u16 f2bf(float x) {           // f32 -> bf16 RNE
  union { float f; unsigned u; } a; a.f = x;
  unsigned r = a.u + 0x7fffu + ((a.u >> 16) & 1u);
  return (u16)(r >> 16);
}
__device__ __forceinline__ float sigmf(float x) { return 1.f / (1.f + expf(-x)); }
__device__ __forceinline__ float gelu_t(float x) {       // jax.nn.gelu (tanh approx)
  float u = 0.7978845608028654f * (x + 0.044715f * x * x * x);
  return 0.5f * x * (1.f + tanhf(u));
}
__device__ __forceinline__ int wrapi(int i, int d) { return i < 0 ? i + d : (i >= d ? i - d : i); }

// copy one feature row (global f32 -> bf16), lanes cooperate, accumulate sumsq
__device__ __forceinline__ void crow(const float* __restrict__ src, int len,
                                     u16* dst, float& ss, int lane) {
  for (int i = lane; i < len; i += 32) {
    float v = src[i];
    ss += v * v;
    dst[i] = f2bf(v);
  }
}

// base + 6 face neighbours of a cubic table (d,d,d,F)
__device__ __forceinline__ int emit3(const float* __restrict__ tbl, int d, int F,
                                     float p0, float p1, float p2,
                                     u16* dst, int off, float& ss, int lane) {
  int i0 = (int)(p0 * (d - 1)), i1 = (int)(p1 * (d - 1)), i2 = (int)(p2 * (d - 1));
  crow(tbl + (size_t)((i0 * d + i1) * d + i2) * F, F, dst + off, ss, lane); off += F;
  const int ox[6] = {1,-1,0,0,0,0}, oy[6] = {0,0,1,-1,0,0}, oz[6] = {0,0,0,0,1,-1};
  for (int f = 0; f < 6; ++f) {
    int a = wrapi(i0 + ox[f], d), b = wrapi(i1 + oy[f], d), c = wrapi(i2 + oz[f], d);
    crow(tbl + (size_t)((a * d + b) * d + c) * F, F, dst + off, ss, lane); off += F;
  }
  return off;
}

// base + 8 ST8 neighbours of a 4D table (d0,d1,d2,d3,F)
__device__ __forceinline__ int emit4(const float* __restrict__ tbl,
                                     int d0, int d1, int d2, int d3, int F,
                                     float p0, float p1, float p2, float tv,
                                     u16* dst, int off, float& ss, int lane) {
  int i0 = (int)(p0 * (d0 - 1)), i1 = (int)(p1 * (d1 - 1));
  int i2 = (int)(p2 * (d2 - 1)), i3 = (int)(tv * (d3 - 1));
  crow(tbl + (size_t)((((i0 * d1 + i1) * d2 + i2) * d3) + i3) * F, F, dst + off, ss, lane); off += F;
  const int O[8][4] = {{1,0,0,0},{-1,0,0,0},{0,1,0,0},{0,-1,0,0},
                       {0,0,1,0},{0,0,-1,0},{0,0,0,1},{0,0,0,-1}};
  for (int f = 0; f < 8; ++f) {
    int a = wrapi(i0 + O[f][0], d0), b = wrapi(i1 + O[f][1], d1);
    int c = wrapi(i2 + O[f][2], d2), e = wrapi(i3 + O[f][3], d3);
    crow(tbl + (size_t)((((a * d1 + b) * d2 + c) * d3) + e) * F, F, dst + off, ss, lane); off += F;
  }
  return off;
}

// ------------------------------------------------- small per-point MLP -> p
__device__ __forceinline__ void rmsn(const float* x, const float* g, float* o, int L) {
  float s = 0.f;
  for (int j = 0; j < L; ++j) s += x[j] * x[j];
  float r = 1.f / (sqrtf(s / (float)L) + EPS);
  for (int j = 0; j < L; ++j) o[j] = g[j] * x[j] * r;
}

__global__ void __launch_bounds__(256)
mlp_kernel(const float* __restrict__ pos, const float* __restrict__ t,
           const float* __restrict__ sph_proj,
           const float* __restrict__ tb1, const float* __restrict__ tb2,
           const float* __restrict__ gn1, const float* __restrict__ gw1,
           const float* __restrict__ gn2, const float* __restrict__ gw2,
           const float* __restrict__ gn3, const float* __restrict__ gw3,
           float* __restrict__ p_out) {
  int i = blockIdx.x * 256 + threadIdx.x;
  if (i >= NPTS) return;
  float px = pos[i*3+0], py = pos[i*3+1], pz = pos[i*3+2];
  float rho = sqrtf(px*px + py*py + pz*pz);
  float phi = atan2f(py, px);
  float ca  = fminf(1.f, fmaxf(-1.f, pz / rho));
  float th  = acosf(ca);
  float h[12];
  for (int j = 0; j < 8; ++j)
    h[j] = rho * sph_proj[j] + phi * sph_proj[8 + j] + th * sph_proj[16 + j];
  float tv = t[i];
  float ct = cosf(tv + tb1[0]), st = sinf(tv + tb2[0]);
  h[8] = ct; h[9] = st; h[10] = ct * sigmf(ct); h[11] = st * sigmf(st);

  float hn[16], y[32], h2[16];
  rmsn(h, gn1, hn, 12);
  for (int o = 0; o < 32; ++o) {
    float s = 0.f;
    for (int j = 0; j < 12; ++j) s += hn[j] * gw1[j * 32 + o];
    y[o] = s;
  }
  for (int k = 0; k < 16; ++k) h2[k] = y[k] * gelu_t(y[16 + k]);

  rmsn(h2, gn2, hn, 16);
  for (int o = 0; o < 32; ++o) {
    float s = 0.f;
    for (int j = 0; j < 16; ++j) s += hn[j] * gw2[j * 32 + o];
    y[o] = s;
  }
  for (int k = 0; k < 16; ++k) h2[k] = y[k] * gelu_t(y[16 + k]);

  rmsn(h2, gn3, hn, 16);
  for (int c = 0; c < 3; ++c) {
    float s = 0.f;
    for (int j = 0; j < 16; ++j) s += hn[j] * gw3[j * 3 + c];
    p_out[i * 3 + c] = sigmf(s);
  }
}

// ------------------------------------- weight prep: fold rmsnorm scale, bf16,
// swizzle into this kernel's B-register order (contiguous k, as A): lane = column n (lanes 16-31 = K+16),
// dword d holds K = h*16 + 2d, 2d+1 within a 32-K tile.
__global__ void __launch_bounds__(256)
prep_w1(const float* __restrict__ vn1, const float* __restrict__ vw1,
        u16* __restrict__ w1p) {
  int idx = blockIdx.x * 256 + threadIdx.x;                 // one packed dword
  const int TOTALD = KT1 * NT1 * 32 * 8;
  if (idx >= TOTALD) return;
  int d = idx & 7, l = (idx >> 3) & 31, tile = idx >> 8;
  int nt = tile % NT1, kt = tile / NT1;
  int h = l >> 4, n = nt * 16 + (l & 15);
  int k = kt * 32 + h * 16 + d * 2;
  float s0 = vn1[k]     * vw1[(size_t)k       * NOUT1 + n];
  float s1 = vn1[k + 1] * vw1[(size_t)(k + 1) * NOUT1 + n];
  ((unsigned*)w1p)[idx] = (unsigned)f2bf(s0) | ((unsigned)f2bf(s1) << 16);
}

__global__ void __launch_bounds__(256)
prep_w2(const float* __restrict__ vn2, const float* __restrict__ vw2,
        u16* __restrict__ w2p) {
  int idx = blockIdx.x * 256 + threadIdx.x;
  const int TOTALD = KT1 * 4 * 32 * 8;
  if (idx >= TOTALD) return;
  int d = idx & 7, l = (idx >> 3) & 31, tile = idx >> 8;
  int nt = tile % 4, kt = tile / 4;
  int h = l >> 4, n = nt * 16 + (l & 15);
  int k = kt * 32 + h * 16 + d * 2;
  float s0 = vn2[k]     * vw2[(size_t)k       * 64 + n];
  float s1 = vn2[k + 1] * vw2[(size_t)(k + 1) * 64 + n];
  ((unsigned*)w2p)[idx] = (unsigned)f2bf(s0) | ((unsigned)f2bf(s1) << 16);
}

// -------------------------------------------- fused gather + double GEMM
__global__ void __launch_bounds__(256)
gemm_fused(const float* __restrict__ p_ws, const float* __restrict__ t,
           const float* __restrict__ g0, const float* __restrict__ g1,
           const float* __restrict__ g2, const float* __restrict__ g3,
           const float* __restrict__ s4a, const float* __restrict__ s4b,
           const float* __restrict__ gtf,
           const u16* __restrict__ w1p, const u16* __restrict__ w2p,
           const float* __restrict__ fw, const float* __restrict__ fb,
           float* __restrict__ out) {
  extern __shared__ char smem[];
  u16*   As  = (u16*)smem;                                   // 64*2336 bf16 = 292KB
  u16*   Vs  = (u16*)(smem + MT * KTOT * 2);                 // 8 waves * 16x32 bf16
  float* rrS = (float*)(smem + MT * KTOT * 2 + 8 * 512 * 2); // 64 row rms of allf

  const int tid = threadIdx.x, lane = tid & 31, wave = tid >> 5;
  const int rowbase = blockIdx.x * MT;

  // ---- fused gather: wave w fills rows [w*8, w*8+8) of the A strip (bf16) ----
  for (int rloc = 0; rloc < 8; ++rloc) {
    const int lrow = wave * 8 + rloc;
    const int grow = rowbase + lrow;
    const float p0 = p_ws[grow*3+0], p1 = p_ws[grow*3+1], p2 = p_ws[grow*3+2];
    const float tv = t[grow];
    u16* dst = As + (size_t)lrow * KTOT;
    float ss = 0.f;
    int off = 0;
    off = emit3(g0, 128, 16, p0, p1, p2, dst, off, ss, lane);
    off = emit3(g1,  64, 32, p0, p1, p2, dst, off, ss, lane);
    off = emit3(g2,  32, 64, p0, p1, p2, dst, off, ss, lane);
    off = emit3(g3,  16, 128, p0, p1, p2, dst, off, ss, lane);
    off = emit4(s4a, 16, 16, 16, 64, 64, p0, p1, p2, tv, dst, off, ss, lane);
    off = emit4(s4b, 64, 64, 32, 16,  8, p0, p1, p2, tv, dst, off, ss, lane);
    {
      int i0 = (int)(p0 * 3.f), i1 = (int)(p1 * 3.f), i2 = (int)(p2 * 3.f);
      int i3 = (int)(tv * 65535.f);
      crow(gtf + ((size_t)((i0 * 4 + i1) * 4 + i2) * 65536 + i3) * 8, 8,
           dst + off, ss, lane);
      off += 8;
    }
    ss += __shfl_xor(ss, 1, 32);  ss += __shfl_xor(ss, 2, 32);
    ss += __shfl_xor(ss, 4, 32);  ss += __shfl_xor(ss, 8, 32);
    ss += __shfl_xor(ss, 16, 32);
    if (lane == 0) rrS[lrow] = 1.f / (sqrtf(ss * (1.f / KTOT)) + EPS);
  }
  __syncthreads();

  // wave split: rg = 16-row group, jh = which half of the 146 v-column tiles
  const int rg = wave & 3, jh = wave >> 2;
  const int m = lane & 15, h = lane >> 4;     // A/B/D lane decomposition
  float rrx[8];
  #pragma unroll
  for (int r = 0; r < 8; ++r) rrx[r] = rrS[rg * 16 + r + 8 * h];

  const u16* Arow = As + (size_t)(rg * 16 + m) * KTOT;  // lane m = matrix row
  u16* myV = Vs + wave * 512;                           // 16x32 bf16 v strip

  v8f dacc[4];
  #pragma unroll
  for (int n2 = 0; n2 < 4; ++n2) dacc[n2] = (v8f){0.f,0.f,0.f,0.f,0.f,0.f,0.f,0.f};
  float ssq[8] = {0.f,0.f,0.f,0.f,0.f,0.f,0.f,0.f};

  const int jt0   = (jh == 0) ? 0 : 72;      // 72/74 split keeps 32-aligned j base
  const int nwin  = (jh == 0) ? 36 : 37;     // windows of 2 jtile-pairs (64 j each? 32 j)

  for (int jw = 0; jw < nwin; ++jw) {
    for (int jp = 0; jp < 2; ++jp) {
      const int jt = jt0 + jw * 2 + jp;      // v column tile 0..145
      v8f acc_a = (v8f){0.f,0.f,0.f,0.f,0.f,0.f,0.f,0.f};
      v8f acc_g = (v8f){0.f,0.f,0.f,0.f,0.f,0.f,0.f,0.f};
      #pragma unroll 2
      for (int kt = 0; kt < KT1; ++kt) {
        BF16Tile a, ba, bg;
        a.q[0]  = *(const u32x4*)(Arow + kt * 32 + h * 16);       // A layout
        a.q[1]  = *(const u32x4*)(Arow + kt * 32 + h * 16 + 8);
        const u16* pb = w1p + ((size_t)(kt * NT1 + jt) * 32 + lane) * 16;
        ba.q[0] = *(const u32x4*)(pb);
        ba.q[1] = *(const u32x4*)(pb + 8);
        const u16* pg = w1p + ((size_t)(kt * NT1 + jt + JT) * 32 + lane) * 16;
        bg.q[0] = *(const u32x4*)(pg);
        bg.q[1] = *(const u32x4*)(pg + 8);
        acc_a = __builtin_amdgcn_wmma_f32_16x16x32_bf16(false, a.v, false, ba.v,
                                                        (short)0, acc_a, false, false);
        acc_g = __builtin_amdgcn_wmma_f32_16x16x32_bf16(false, a.v, false, bg.v,
                                                        (short)0, acc_g, false, false);
      }
      // geglu epilogue; deferred per-row 1/(rms+eps) of allf applies linearly
      #pragma unroll
      for (int r = 0; r < 8; ++r) {
        float va = acc_a[r] * rrx[r];
        float vg = acc_g[r] * rrx[r];
        float v  = va * gelu_t(vg);
        ssq[r] += v * v;
        myV[(r + 8 * h) * 32 + jp * 16 + m] = f2bf(v);   // lane m = column here
      }
    }
    // fold this 32-wide v window into `down` with a second bf16 WMMA chain
    const int kt2 = (jt0 >> 1) + jw;                     // 32-aligned j/32 tile
    BF16Tile av;
    av.q[0] = *(const u32x4*)(myV + m * 32 + h * 16);
    av.q[1] = *(const u32x4*)(myV + m * 32 + h * 16 + 8);
    #pragma unroll
    for (int n2 = 0; n2 < 4; ++n2) {
      BF16Tile b2;
      const u16* pb = w2p + ((size_t)(kt2 * 4 + n2) * 32 + lane) * 16;
      b2.q[0] = *(const u32x4*)(pb);
      b2.q[1] = *(const u32x4*)(pb + 8);
      dacc[n2] = __builtin_amdgcn_wmma_f32_16x16x32_bf16(false, av.v, false, b2.v,
                                                         (short)0, dacc[n2], false, false);
    }
  }

  // per-row sumsq of v: reduce over the 16 column lanes
  #pragma unroll
  for (int r = 0; r < 8; ++r) {
    float s = ssq[r];
    s += __shfl_xor(s, 1, 32); s += __shfl_xor(s, 2, 32);
    s += __shfl_xor(s, 4, 32); s += __shfl_xor(s, 8, 32);
    ssq[r] = s;
  }
  __syncthreads();                       // A strip no longer needed; reuse as scratch
  float* red = (float*)smem;             // 8 waves * 16 rows * 64 cols
  float* ssr = red + 8 * 1024;           // 8 waves * 16 rows
  #pragma unroll
  for (int n2 = 0; n2 < 4; ++n2)
    #pragma unroll
    for (int r = 0; r < 8; ++r)
      red[wave * 1024 + (r + 8 * h) * 64 + n2 * 16 + m] = dacc[n2][r];
  if (m == 0)
    #pragma unroll
    for (int r = 0; r < 8; ++r) ssr[wave * 16 + r + 8 * h] = ssq[r];
  __syncthreads();

  if (jh == 0) {   // waves 0..3: combine halves, rmsnorm(v), final 68->4 linear
    const int row  = m;
    const int grow = rowbase + rg * 16 + row;
    float sst = ssr[wave * 16 + row] + ssr[(wave + 4) * 16 + row];
    float rrv = 1.f / (sqrtf(sst * (1.f / KTOT)) + EPS);
    float p0 = p_ws[grow*3+0], p1 = p_ws[grow*3+1], p2 = p_ws[grow*3+2];
    float tv = t[grow];
    for (int cc = 0; cc < 2; ++cc) {
      int c = h * 2 + cc;
      float acc = fb[c] + p0 * fw[64*4+c] + p1 * fw[65*4+c]
                        + p2 * fw[66*4+c] + tv * fw[67*4+c];
      for (int i = 0; i < 64; ++i) {
        float dn = (red[wave * 1024 + row * 64 + i] +
                    red[(wave + 4) * 1024 + row * 64 + i]) * rrv;
        acc += dn * fw[i * 4 + c];
      }
      out[grow * 4 + c] = acc;
    }
  }
}

// ----------------------------------------------------------------- launcher
extern "C" void kernel_launch(void* const* d_in, const int* in_sizes, int n_in,
                              void* d_out, int out_size, void* d_ws, size_t ws_size,
                              hipStream_t stream) {
  const float* pos  = (const float*)d_in[0];
  const float* t    = (const float*)d_in[2];
  const float* tbl0 = (const float*)d_in[3];
  const float* tbl1 = (const float*)d_in[4];
  const float* tbl2 = (const float*)d_in[5];
  const float* tbl3 = (const float*)d_in[6];
  const float* st1  = (const float*)d_in[7];
  const float* st2  = (const float*)d_in[8];
  const float* tf   = (const float*)d_in[9];
  const float* sph  = (const float*)d_in[10];
  const float* tb1  = (const float*)d_in[11];
  const float* tb2  = (const float*)d_in[12];
  const float* gn1  = (const float*)d_in[13];
  const float* gw1  = (const float*)d_in[14];
  const float* gn2  = (const float*)d_in[15];
  const float* gw2  = (const float*)d_in[16];
  const float* gn3  = (const float*)d_in[17];
  const float* gw3  = (const float*)d_in[18];
  const float* vn1  = (const float*)d_in[19];
  const float* vw1  = (const float*)d_in[20];
  const float* vn2  = (const float*)d_in[21];
  const float* vw2  = (const float*)d_in[22];
  const float* fw   = (const float*)d_in[23];
  const float* fb   = (const float*)d_in[24];
  float* out = (float*)d_out;

  char* ws = (char*)d_ws;
  float* p_ws = (float*)ws;                 ws += (size_t)NPTS * 3 * 4;      // 393216
  u16*   w1p  = (u16*)ws;                   ws += (size_t)KTOT * NOUT1 * 2;  // 21.8MB
  u16*   w2p  = (u16*)ws;                   // 2336*64*2

  mlp_kernel<<<NPTS / 256, 256, 0, stream>>>(pos, t, sph, tb1, tb2,
                                             gn1, gw1, gn2, gw2, gn3, gw3, p_ws);
  prep_w1<<<(KT1 * NT1 * 32 * 8) / 256, 256, 0, stream>>>(vn1, vw1, w1p);
  prep_w2<<<(KT1 * 4 * 32 * 8 + 255) / 256, 256, 0, stream>>>(vn2, vw2, w2p);

  const size_t smem = (size_t)MT * KTOT * 2 + 8 * 512 * 2 + 64 * 4; // 307456 B
  gemm_fused<<<NPTS / MT, 256, smem, stream>>>(p_ws, t,
                                               tbl0, tbl1, tbl2, tbl3,
                                               st1, st2, tf,
                                               w1p, w2p, fw, fb, out);
}
